// voxel_dgcnn_48026324304117
// MI455X (gfx1250) — matrix operand order repaired and checked by reading, not compiled, not run on hardware
//
#include <hip/hip_runtime.h>
#include <hip/hip_bf16.h>

typedef __attribute__((ext_vector_type(16))) _Float16     v16h;
typedef __attribute__((ext_vector_type(8)))  float        v8f;
typedef __attribute__((ext_vector_type(4)))  unsigned int v4u;
typedef __attribute__((ext_vector_type(4)))  int          v4i;
typedef __attribute__((ext_vector_type(8)))  int          v8i;

#define NPTS 200       // points per voxel
#define NP   208       // padded to 13 row-tiles of 16
#define KNN  10
#define NVOX 512
#define EPSB 1e-5f
#define BIGF 1e7f

__device__ __forceinline__ float leakyf(float z) { return z >= 0.f ? z : 0.2f * z; }

// ---- WMMA fragment index helpers (CDNA5 ISA 7.12.2, wave32) ----
// A (16x32 f16): lane = 16*half + row; element e -> K = e + 8*half + (e>=8 ? 8 : 0)
// B (32x16 f16): lane = 16*half + col; element e -> K = e + 8*half + (e>=8 ? 8 : 0)
// C/D (16x16 f32): lane = 16*half + col; vgpr v -> row M = v + 8*half
__device__ __forceinline__ int kA(int e, int half) { return e + 8 * half + (e >= 8 ? 8 : 0); }
__device__ __forceinline__ int kB(int e, int half) { return e + 8 * half + (e >= 8 ? 8 : 0); }

// =====================================================================
// Tensor Data Mover: 1-D f32 block copy global -> LDS.
// D# per CDNA5 ISA:
//  group0: [1:0]=count=1, [63:32]=lds_addr, [120:64]=global_addr, [127:126]=type=2
//  group1: [17:16]=data_size=2 (4B), [79:48]=tensor_dim0, [127:112]=tile_dim0
//  groups 2/3: zero (<=2D tensor)
// =====================================================================
__device__ __forceinline__ void tdm_load_f32(unsigned lds_byte_off,
                                             const void* gptr, unsigned nelem) {
    unsigned long long ga = (unsigned long long)(size_t)gptr;
    v4u g0;
    g0[0] = 1u;                                            // count=1, user mode
    g0[1] = lds_byte_off;                                  // lds_addr
    g0[2] = (unsigned)(ga & 0xffffffffu);                  // global_addr[31:0]
    g0[3] = (unsigned)((ga >> 32) & 0x01ffffffu) | (2u << 30); // addr[56:32] | type=2
    v8i g1 = {};
    g1[0] = (int)(2u << 16);                               // data_size = 4 bytes
    g1[1] = (int)((nelem & 0xffffu) << 16);                // tensor_dim0[15:0]
    g1[2] = (int)((nelem >> 16) & 0xffffu);                // tensor_dim0[31:16]
    g1[3] = (int)((nelem & 0xffffu) << 16);                // tile_dim0
    g1[5] = (int)nelem;                                    // tensor_dim0_stride[31:0]
    v4i gz = {};
#if __has_include(<hip/amd_detail/amd_gfx1250_TDM.h>)
    v8i gz8 = {};
    __builtin_amdgcn_tensor_load_to_lds(g0, g1, gz, gz, gz8, 0);   // therock 6-arg
#else
    __builtin_amdgcn_tensor_load_to_lds(g0, g1, gz, gz, 0);        // ROCm 7.2 5-arg
#endif
}

// =====================================================================
// Fused edge-conv layer: per-voxel workgroup (256 thr = 8 waves).
//   P = X * Wd^T ; Q = X * (Wc - Wd)^T   (WMMA f16->f32)
//   knn (f32 VALU) ; out[n,o] = leaky( max_k P[idx[n,k],o] + Q[n,o] )
// C = real channels, CP = padded (mult of 32), O = out channels (mult 16)
// =====================================================================
template <int C, int CP, int O>
__global__ __launch_bounds__(256) void edge_layer_kernel(
    const float* __restrict__ xin,   // (NVOX, NPTS, C)
    const float* __restrict__ W,     // (O, 2C)
    float* __restrict__ xout)        // (NVOX, NPTS, O)
{
    extern __shared__ char smem_raw[];
    float*    Xf   = reinterpret_cast<float*>(smem_raw);        // NP*CP f32 (LDS offset 0)
    float*    xx   = Xf + NP * CP;                              // NP f32
    int*      idxS = reinterpret_cast<int*>(xx + NP);           // NPTS*KNN
    _Float16* Xh   = reinterpret_cast<_Float16*>(idxS + NPTS * KNN); // NP*CP
    _Float16* WdT  = Xh + NP * CP;                              // CP*O  [k][o]
    _Float16* WqT  = WdT + CP * O;                              // CP*O
    _Float16* Ph   = WqT + CP * O;                              // NP*O
    _Float16* Qh   = Ph + NP * O;                               // NP*O

    const int b    = blockIdx.x;
    const int tid  = threadIdx.x;
    const int lane = tid & 31;
    const int wv   = tid >> 5;
    const int half = lane >> 4;
    const int lm   = lane & 15;

    // ---- W split: WdT[k][o] = W[o][k] ; WqT[k][o] = W[o][C+k]-W[o][k] ----
    for (int i = tid; i < CP * O; i += 256) {
        int k = i / O, o = i - k * O;
        float wd = (k < C) ? W[o * (2 * C) + k] : 0.f;
        float wc = (k < C) ? W[o * (2 * C) + C + k] : 0.f;
        WdT[i] = (_Float16)wd;
        WqT[i] = (_Float16)(wc - wd);
    }

    // ---- stage X: TDM path when the voxel slab is contiguous (C==CP) ----
    if constexpr (C == CP) {
        for (int i = NPTS * CP + tid; i < NP * CP; i += 256) Xf[i] = 0.f; // pad rows
        if (tid == 0) {
            tdm_load_f32(0u, xin + (size_t)b * NPTS * C, NPTS * C);
            __builtin_amdgcn_s_wait_tensorcnt((short)0);
        }
        __syncthreads();
        for (int i = tid; i < NP * CP; i += 256) Xh[i] = (_Float16)Xf[i];
    } else {
        for (int i = tid; i < NP * CP; i += 256) {
            int n = i / CP, c = i - n * CP;
            float v = (n < NPTS && c < C) ? xin[(size_t)b * NPTS * C + n * C + c] : 0.f;
            Xf[i] = v;
            Xh[i] = (_Float16)v;
        }
    }
    __syncthreads();

    // ---- point norms ----
    for (int n = tid; n < NP; n += 256) {
        float s = 0.f;
#pragma unroll
        for (int c = 0; c < C; ++c) { float v = Xf[n * CP + c]; s += v * v; }
        xx[n] = s;
    }
    __syncthreads();

    // ---- P / Q GEMMs via WMMA ----
    constexpr int RT = NP / 16, CT = O / 16, KS = CP / 32;
    for (int t = wv; t < RT * CT * 2; t += 8) {
        const int which = (t >= RT * CT);
        const int tt = which ? (t - RT * CT) : t;
        const int rt = tt / CT, ct = tt - rt * CT;
        const _Float16* Bsrc = which ? WqT : WdT;
        v8f acc = {};
        const int arow = rt * 16 + lm;
#pragma unroll
        for (int ks = 0; ks < KS; ++ks) {
            v16h a, bf;
#pragma unroll
            for (int e = 0; e < 16; ++e) a[e]  = Xh[arow * CP + ks * 32 + kA(e, half)];
#pragma unroll
            for (int e = 0; e < 16; ++e) bf[e] = Bsrc[(ks * 32 + kB(e, half)) * O + ct * 16 + lm];
            acc = __builtin_amdgcn_wmma_f32_16x16x32_f16(false, a, false, bf,
                                                         (short)0, acc, false, false);
        }
        _Float16* Dst = which ? Qh : Ph;
#pragma unroll
        for (int v = 0; v < 8; ++v) {
            int row = rt * 16 + v + 8 * half;
            Dst[row * O + ct * 16 + lm] = (_Float16)acc[v];
        }
    }

    // ---- brute-force KNN (one point per thread, f32) ----
    if (tid < NPTS) {
        float xi[CP];
#pragma unroll
        for (int c = 0; c < C; ++c) xi[c] = Xf[tid * CP + c];
        const float xxi = xx[tid];
        float best[KNN];
        int   bidx[KNN];
#pragma unroll
        for (int k = 0; k < KNN; ++k) { best[k] = -3.0e38f; bidx[k] = 0; }
        for (int j = 0; j < NPTS; ++j) {
            float dot = 0.f;
#pragma unroll
            for (int c = 0; c < C; ++c) dot += xi[c] * Xf[j * CP + c];
            float pd = 2.f * dot - xxi - xx[j];
            pd *= (xx[j] > 0.f) ? 1.f : BIGF;
            if (pd > best[KNN - 1]) {
                int p = KNN - 1;
                while (p > 0 && best[p - 1] < pd) {
                    best[p] = best[p - 1]; bidx[p] = bidx[p - 1]; --p;
                }
                best[p] = pd; bidx[p] = j;
            }
        }
#pragma unroll
        for (int k = 0; k < KNN; ++k) idxS[tid * KNN + k] = bidx[k];
    }
    __syncthreads();

    // ---- gather + max over K + leaky ----
    for (int e = tid; e < NPTS * O; e += 256) {
        int n = e / O, o = e - n * O;
        float q = (float)Qh[n * O + o];
        float m = -3.0e38f;
#pragma unroll
        for (int k = 0; k < KNN; ++k)
            m = fmaxf(m, (float)Ph[idxS[n * KNN + k] * O + o]);
        xout[(size_t)b * NPTS * O + e] = leakyf(m + q);
    }
}

// =====================================================================
// W5 stage + BN + leaky + max/mean pooling. Per-voxel workgroup.
// xc = concat(x1,x2,x3,x4) (NPTS x 256) -> y = leaky(bn(xc*W5^T)) (NPTS x 512)
// pooled[b] = [max_n y | mean_n y]  (1024)
// =====================================================================
__global__ __launch_bounds__(256) void w5_pool_kernel(
    const float* __restrict__ x1, const float* __restrict__ x2,
    const float* __restrict__ x3, const float* __restrict__ x4,
    const float* __restrict__ W5,                     // (512,256)
    const float* __restrict__ g5, const float* __restrict__ b5,
    const float* __restrict__ m5, const float* __restrict__ v5,
    float* __restrict__ pooled)                       // (NVOX,1024)
{
    extern __shared__ char smem_raw[];
    _Float16* XCh = reinterpret_cast<_Float16*>(smem_raw);   // NP x 256

    const int b    = blockIdx.x;
    const int tid  = threadIdx.x;
    const int lane = tid & 31;
    const int wv   = tid >> 5;
    const int half = lane >> 4;
    const int lm   = lane & 15;

    for (int i = tid; i < NP * 256; i += 256) {
        int n = i >> 8, c = i & 255;
        float v = 0.f;
        if (n < NPTS) {
            if (c < 32)       v = x1[(size_t)b * NPTS * 32 + n * 32 + c];
            else if (c < 64)  v = x2[(size_t)b * NPTS * 32 + n * 32 + (c - 32)];
            else if (c < 128) v = x3[(size_t)b * NPTS * 64 + n * 64 + (c - 64)];
            else              v = x4[(size_t)b * NPTS * 128 + n * 128 + (c - 128)];
        }
        XCh[i] = (_Float16)v;
    }
    __syncthreads();

    for (int ct = wv; ct < 32; ct += 8) {            // 512/16 col tiles
        const int ch = ct * 16 + lm;
        const float sc = g5[ch] * rsqrtf(v5[ch] + EPSB);
        const float bs = b5[ch] - m5[ch] * sc;
        float mx = -3.0e38f, sm = 0.f;
        // 12 full row tiles (rows 0..191): no masking, no exec churn
        for (int rt = 0; rt < 12; ++rt) {
            v8f acc = {};
            const int arow = rt * 16 + lm;
#pragma unroll
            for (int ks = 0; ks < 8; ++ks) {          // 256/32
                v16h a, bf;
#pragma unroll
                for (int e = 0; e < 16; ++e) a[e] = XCh[arow * 256 + ks * 32 + kA(e, half)];
#pragma unroll
                for (int e = 0; e < 16; ++e)
                    bf[e] = (_Float16)W5[(size_t)ch * 256 + ks * 32 + kB(e, half)];
                acc = __builtin_amdgcn_wmma_f32_16x16x32_f16(false, a, false, bf,
                                                             (short)0, acc, false, false);
            }
#pragma unroll
            for (int v = 0; v < 8; ++v) {
                float y = leakyf(acc[v] * sc + bs);
                mx = fmaxf(mx, y);
                sm += y;
            }
        }
        // masked tail tile (rows 192..207; valid up to 199)
        {
            v8f acc = {};
            const int arow = 12 * 16 + lm;
#pragma unroll
            for (int ks = 0; ks < 8; ++ks) {
                v16h a, bf;
#pragma unroll
                for (int e = 0; e < 16; ++e) a[e] = XCh[arow * 256 + ks * 32 + kA(e, half)];
#pragma unroll
                for (int e = 0; e < 16; ++e)
                    bf[e] = (_Float16)W5[(size_t)ch * 256 + ks * 32 + kB(e, half)];
                acc = __builtin_amdgcn_wmma_f32_16x16x32_f16(false, a, false, bf,
                                                             (short)0, acc, false, false);
            }
#pragma unroll
            for (int v = 0; v < 8; ++v) {
                int n = 12 * 16 + v + 8 * half;
                bool ok = (n < NPTS);
                float y = leakyf(acc[v] * sc + bs);
                mx = fmaxf(mx, ok ? y : -3.0e38f);
                sm += ok ? y : 0.f;
            }
        }
        // combine the two half-wave partials (cols repeat at lane+16)
        mx = fmaxf(mx, __shfl_xor(mx, 16, 32));
        sm += __shfl_xor(sm, 16, 32);
        if (lane < 16) {
            pooled[(size_t)b * 1024 + ch]       = mx;
            pooled[(size_t)b * 1024 + 512 + ch] = sm * (1.f / (float)NPTS);
        }
    }
}

// =====================================================================
// FC1: h = leaky(bn6(pooled @ Wl1^T))  (512x1024)x(1024x1024)
// One 16x16 tile per wave. grid 256 x 256thr -> 2048 waves.
// =====================================================================
__global__ __launch_bounds__(256) void fc1_kernel(
    const float* __restrict__ A,     // pooled (512,1024) f32
    const float* __restrict__ Wl1,   // (1024,1024)
    const float* __restrict__ g6, const float* __restrict__ b6,
    const float* __restrict__ m6, const float* __restrict__ v6,
    float* __restrict__ H)           // (512,1024)
{
    const int lane = threadIdx.x & 31;
    const int wv   = threadIdx.x >> 5;
    const int half = lane >> 4;
    const int lm   = lane & 15;

    for (int t = blockIdx.x * 8 + wv; t < 32 * 64; t += gridDim.x * 8) {
        const int r = t >> 6, c = t & 63;
        const int arow = r * 16 + lm;
        const int ocol = c * 16 + lm;
        v8f acc = {};
        for (int ks = 0; ks < 32; ++ks) {            // 1024/32
            if (ks + 1 < 32) {                       // stream-prefetch next W tile
                __builtin_prefetch(&Wl1[(size_t)ocol * 1024 + (ks + 1) * 32], 0, 3);
            }
            v16h a, bf;
#pragma unroll
            for (int e = 0; e < 16; ++e)
                a[e] = (_Float16)A[(size_t)arow * 1024 + ks * 32 + kA(e, half)];
#pragma unroll
            for (int e = 0; e < 16; ++e)
                bf[e] = (_Float16)Wl1[(size_t)ocol * 1024 + ks * 32 + kB(e, half)];
            acc = __builtin_amdgcn_wmma_f32_16x16x32_f16(false, a, false, bf,
                                                         (short)0, acc, false, false);
        }
        const float sc = g6[ocol] * rsqrtf(v6[ocol] + EPSB);
        const float bs = b6[ocol] - m6[ocol] * sc;
#pragma unroll
        for (int v = 0; v < 8; ++v) {
            int row = r * 16 + v + 8 * half;
            H[(size_t)row * 1024 + ocol] = leakyf(acc[v] * sc + bs);
        }
    }
}

// =====================================================================
// FC2: out = leaky(bn7(h @ Wl2^T + bl2))  (512x1024)x(1024x40) -> (512,40)
// 3 col tiles (48 padded, masked at 40). grid 12 x 256thr -> 96 waves.
// =====================================================================
__global__ __launch_bounds__(256) void fc2_kernel(
    const float* __restrict__ H,     // (512,1024)
    const float* __restrict__ Wl2,   // (40,1024)
    const float* __restrict__ bl2,
    const float* __restrict__ g7, const float* __restrict__ b7,
    const float* __restrict__ m7, const float* __restrict__ v7,
    float* __restrict__ out)         // (512,40)
{
    const int lane = threadIdx.x & 31;
    const int wv   = threadIdx.x >> 5;
    const int half = lane >> 4;
    const int lm   = lane & 15;

    for (int t = blockIdx.x * 8 + wv; t < 32 * 3; t += gridDim.x * 8) {
        const int r = t / 3, c = t - r * 3;
        const int arow = r * 16 + lm;
        const int ocol = c * 16 + lm;
        const bool valid = (ocol < 40);
        v8f acc = {};
        for (int ks = 0; ks < 32; ++ks) {
            v16h a, bf;
#pragma unroll
            for (int e = 0; e < 16; ++e)
                a[e] = (_Float16)H[(size_t)arow * 1024 + ks * 32 + kA(e, half)];
#pragma unroll
            for (int e = 0; e < 16; ++e) {
                float w = valid ? Wl2[(size_t)ocol * 1024 + ks * 32 + kB(e, half)] : 0.f;
                bf[e] = (_Float16)w;
            }
            acc = __builtin_amdgcn_wmma_f32_16x16x32_f16(false, a, false, bf,
                                                         (short)0, acc, false, false);
        }
        if (valid) {
            const float sc = g7[ocol] * rsqrtf(v7[ocol] + EPSB);
            const float bs = b7[ocol] - m7[ocol] * sc;
            const float bias = bl2[ocol];
#pragma unroll
            for (int v = 0; v < 8; ++v) {
                int row = r * 16 + v + 8 * half;
                out[(size_t)row * 40 + ocol] = leakyf((acc[v] + bias) * sc + bs);
            }
        }
    }
}

// =====================================================================
// Host launcher
// =====================================================================
static constexpr size_t edge_smem_bytes(int CP, int O) {
    return (size_t)NP * CP * 4         // Xf
         + (size_t)NP * 4              // xx
         + (size_t)NPTS * KNN * 4      // idx
         + (size_t)NP * CP * 2         // Xh
         + 2ull * CP * O * 2           // WdT, WqT
         + 2ull * NP * O * 2;          // P, Q
}

extern "C" void kernel_launch(void* const* d_in, const int* in_sizes, int n_in,
                              void* d_out, int out_size, void* d_ws, size_t ws_size,
                              hipStream_t stream) {
    const float* x   = (const float*)d_in[0];
    const float* W1  = (const float*)d_in[1];
    const float* W2  = (const float*)d_in[2];
    const float* W3  = (const float*)d_in[3];
    const float* W4  = (const float*)d_in[4];
    const float* W5  = (const float*)d_in[5];
    const float* g5  = (const float*)d_in[6];
    const float* b5  = (const float*)d_in[7];
    const float* m5  = (const float*)d_in[8];
    const float* v5  = (const float*)d_in[9];
    const float* Wl1 = (const float*)d_in[10];
    const float* g6  = (const float*)d_in[11];
    const float* b6  = (const float*)d_in[12];
    const float* m6  = (const float*)d_in[13];
    const float* v6  = (const float*)d_in[14];
    const float* Wl2 = (const float*)d_in[15];
    const float* bl2 = (const float*)d_in[16];
    const float* g7  = (const float*)d_in[17];
    const float* b7  = (const float*)d_in[18];
    const float* m7  = (const float*)d_in[19];
    const float* v7  = (const float*)d_in[20];

    float* ws = (float*)d_ws;
    const size_t BN = (size_t)NVOX * NPTS;
    float* x1     = ws;                  // BN*32
    float* x2     = x1 + BN * 32;        // BN*32
    float* x3     = x2 + BN * 32;        // BN*64
    float* x4     = x3 + BN * 64;        // BN*128
    float* pooled = x4 + BN * 128;       // NVOX*1024
    float* h      = pooled + (size_t)NVOX * 1024;  // NVOX*1024

    edge_layer_kernel<3, 32, 32><<<NVOX, 256, edge_smem_bytes(32, 32), stream>>>(x, W1, x1);
    edge_layer_kernel<32, 32, 32><<<NVOX, 256, edge_smem_bytes(32, 32), stream>>>(x1, W2, x2);
    edge_layer_kernel<32, 32, 64><<<NVOX, 256, edge_smem_bytes(32, 64), stream>>>(x2, W3, x3);
    edge_layer_kernel<64, 64, 128><<<NVOX, 256, edge_smem_bytes(64, 128), stream>>>(x3, W4, x4);

    w5_pool_kernel<<<NVOX, 256, (size_t)NP * 256 * 2, stream>>>(
        x1, x2, x3, x4, W5, g5, b5, m5, v5, pooled);

    fc1_kernel<<<256, 256, 0, stream>>>(pooled, Wl1, g6, b6, m6, v6, h);
    fc2_kernel<<<12, 256, 0, stream>>>(h, Wl2, bl2, g7, b7, m7, v7, (float*)d_out);
}
